// MyTemporalAttention_81131932222090
// MI455X (gfx1250) — matrix operand order repaired and checked by reading, not compiled, not run on hardware
//
#include <hip/hip_runtime.h>
#include <hip/hip_bf16.h>
#include <stdint.h>

// MyTemporalAttention on MI455X (gfx1250), wave32 + v_wmma_f32_16x16x32_f16.
//
// Shapes: B=32, C=64, L=2048, D_kq=50 (padded to 64 for WMMA K-dim).
// Softmax is over axis=1 (the l axis): M[m],Z[m] are column statistics,
// y[l,c] = sum_m exp(s[l,m]-M[m]) * (v[m,c]/Z[m]).
// Two-pass recompute of s avoids the 512MB s materialization (~1GB HBM
// traffic ~ 44us at 23.3TB/s) in exchange for +13 GFLOP of f16 WMMA.
// All operands stay L2-resident (per-batch K/Q/Vpt = 256KB each << 192MB).

#define B_   32
#define C_   64
#define L_   2048
#define DKQ  50
#define DPAD 64

typedef __attribute__((ext_vector_type(16))) _Float16 v16h;
typedef __attribute__((ext_vector_type(8)))  _Float16 v8h;
typedef __attribute__((ext_vector_type(8)))  float    v8f;

#define WMMA_F16(A, Bm, Cm) \
  __builtin_amdgcn_wmma_f32_16x16x32_f16(false, (A), false, (Bm), (short)0, (Cm), false, false)

// A-matrix 16x32 f16 (ISA 7.12.2): lane holds row (lane&15); K chunks
// [koff..koff+7] and [koff+16..koff+23], koff = (lane>=16 ? 8 : 0).
static __device__ __forceinline__ v16h load_tileA(const _Float16* base, int ld) {
  int lane = threadIdx.x & 31;
  const _Float16* p = base + (lane & 15) * ld + ((lane >> 4) << 3);
  v8h c0 = *(const v8h*)(p);
  v8h c1 = *(const v8h*)(p + 16);
  v16h a;
#pragma unroll
  for (int i = 0; i < 8; ++i) { a[i] = c0[i]; a[i + 8] = c1[i]; }
  return a;
}

// B-matrix 32x16 f16: lane holds column n=(lane&15); K chunks
// [koff..koff+7] and [koff+16..koff+23], koff = (lane>=16 ? 8 : 0). Memory layout: row index = n, K contiguous.
static __device__ __forceinline__ v16h load_tileB(const _Float16* base, int ld) {
  int lane = threadIdx.x & 31;
  const _Float16* p = base + (lane & 15) * ld + ((lane >> 4) << 3);
  v8h c0 = *(const v8h*)(p);
  v8h c1 = *(const v8h*)(p + 16);
  v16h b;
#pragma unroll
  for (int i = 0; i < 8; ++i) { b[i] = c0[i]; b[i + 8] = c1[i]; }
  return b;
}

// ---------------------------------------------------------------------------
// Kernel 1: projections. xt[b,l,:] @ {Wk,Wq,Wv} (+bias, elu/elu/tanh).
// K is pre-scaled by 1/sqrt(sample_len); K,Q padded to 64 cols with zeros.
// ---------------------------------------------------------------------------
__global__ __launch_bounds__(256) void proj_kernel(
    const float* __restrict__ x,  const float* __restrict__ Wk,
    const float* __restrict__ bk, const float* __restrict__ Wq,
    const float* __restrict__ bq, const float* __restrict__ Wv,
    const float* __restrict__ bv, const int* __restrict__ sample_len,
    _Float16* __restrict__ Kh, _Float16* __restrict__ Qh, float* __restrict__ Vf)
{
  __shared__ float sWk[C_ * DKQ], sWq[C_ * DKQ], sWv[C_ * C_];
  __shared__ float sbk[DKQ], sbq[DKQ], sbv[C_];
  int tid = threadIdx.x;
  for (int i = tid; i < C_ * DKQ; i += 256) { sWk[i] = Wk[i]; sWq[i] = Wq[i]; }
  for (int i = tid; i < C_ * C_;  i += 256) sWv[i] = Wv[i];
  if (tid < DKQ) { sbk[tid] = bk[tid]; sbq[tid] = bq[tid]; }
  if (tid < C_)  sbv[tid] = bv[tid];
  __syncthreads();

  int b = blockIdx.y;
  int l = blockIdx.x * 256 + tid;
  float scale = rsqrtf((float)sample_len[0]);

  float xr[C_];
#pragma unroll
  for (int c = 0; c < C_; ++c) xr[c] = x[(b * C_ + c) * L_ + l];  // coalesced over l

  int ro = (b * L_ + l) * DPAD;
  for (int j = 0; j < DKQ; ++j) {
    float ak = sbk[j], aq = sbq[j];
#pragma unroll
    for (int c = 0; c < C_; ++c) {
      ak = fmaf(xr[c], sWk[c * DKQ + j], ak);
      aq = fmaf(xr[c], sWq[c * DKQ + j], aq);
    }
    ak = (ak > 0.f) ? ak : (__expf(ak) - 1.f);   // elu
    aq = (aq > 0.f) ? aq : (__expf(aq) - 1.f);
    Kh[ro + j] = (_Float16)(ak * scale);
    Qh[ro + j] = (_Float16)aq;
  }
  for (int j = DKQ; j < DPAD; ++j) { Kh[ro + j] = (_Float16)0.f; Qh[ro + j] = (_Float16)0.f; }
  for (int j = 0; j < C_; ++j) {
    float av = sbv[j];
#pragma unroll
    for (int c = 0; c < C_; ++c) av = fmaf(xr[c], sWv[c * C_ + j], av);
    Vf[ro + j] = tanhf(av);
  }
}

// ---------------------------------------------------------------------------
// Kernel 2: column softmax stats. One wave owns 16 m-columns, sweeps all l
// with WMMA tiles, keeping an online (max, rescaled-sum) per column.
// A-tiles are prefetched one iteration ahead; the two K-half WMMAs use
// independent accumulators (no D->C chain stall) combined with v_add_f32.
// Emits M[b][m] and Vpt[b][c][m] = tanh_v[m][c] / Z[m] (f16, transposed so
// kernel 3 B-tiles are contiguous loads).
// ---------------------------------------------------------------------------
__global__ __launch_bounds__(256) void stats_kernel(
    const _Float16* __restrict__ Kh, const _Float16* __restrict__ Qh,
    const float* __restrict__ Vf, float* __restrict__ Mc, _Float16* __restrict__ Vpt)
{
  int b    = blockIdx.y;
  int wave = threadIdx.x >> 5;
  int lane = threadIdx.x & 31;
  int m0   = blockIdx.x * 128 + wave * 16;

  const _Float16* Qb = Qh + (size_t)(b * L_ + m0) * DPAD;
  v16h Bq0 = load_tileB(Qb, DPAD);        // d = 0..31  (B = Q^T)
  v16h Bq1 = load_tileB(Qb + 32, DPAD);   // d = 32..63

  const _Float16* Kb = Kh + (size_t)(b * L_) * DPAD;
  float runM = -INFINITY, runZ = 0.f;

  v16h A0 = load_tileA(Kb, DPAD);
  v16h A1 = load_tileA(Kb + 32, DPAD);
  for (int l0 = 0; l0 < L_; l0 += 16) {
    int ln = (l0 + 16) & (L_ - 1);        // wraps to 0 on last iter (dead value)
    v16h nA0 = load_tileA(Kb + ln * DPAD, DPAD);
    v16h nA1 = load_tileA(Kb + ln * DPAD + 32, DPAD);

    v8f Da = {}, Db = {};
    Da = WMMA_F16(A0, Bq0, Da);           // independent -> back-to-back issue
    Db = WMMA_F16(A1, Bq1, Db);
    v8f D = Da + Db;

    float pm = fmaxf(D[0], D[1]);
#pragma unroll
    for (int i = 2; i < 8; ++i) pm = fmaxf(pm, D[i]);
    pm = fmaxf(pm, __shfl_xor(pm, 16, 32));         // full 16-row column max
    float nM = fmaxf(runM, pm);
    float t = 0.f;
#pragma unroll
    for (int i = 0; i < 8; ++i) t += __expf(D[i] - nM);
    t += __shfl_xor(t, 16, 32);
    runZ = runZ * __expf(runM - nM) + t;            // exp(-inf)=0 on first iter
    runM = nM;
    A0 = nA0; A1 = nA1;
  }

  if (lane < 16) Mc[b * L_ + m0 + lane] = runM;

  // Fold 1/Z into v, write transposed f16: Vpt[b][c][m0+m'].
  float rZ = 1.f / runZ;                  // per-lane column m' = lane&15 (dup in both halves)
  int mp = lane & 15;
  const float* Vb = Vf + (size_t)(b * L_ + m0 + mp) * DPAD;
  _Float16* Vo = Vpt + (size_t)b * C_ * L_ + m0 + mp;
#pragma unroll 4
  for (int i = 0; i < 32; ++i) {
    int c = 2 * i + (lane >> 4);          // lanes<16 cover even c, lanes>=16 odd c
    Vo[(size_t)c * L_] = (_Float16)(Vb[c] * rZ);
  }
}

// ---------------------------------------------------------------------------
// Kernel 3: output. One wave owns 16 output rows (l). Recompute s tiles,
// E = exp(s - M[m]) -> f16 via 1KB per-wave LDS bounce (D-layout -> A-layout),
// then 4 accumulating WMMAs vs Vpt per 32-m chunk. 512 WMMAs / wave.
// Q B-tiles prefetched one m-chunk ahead; Vpt B-tiles and column-max loads
// issued before the LDS fence so every global load overlaps compute.
// ---------------------------------------------------------------------------
__global__ __launch_bounds__(256) void attn_kernel(
    const _Float16* __restrict__ Kh, const _Float16* __restrict__ Qh,
    const _Float16* __restrict__ Vpt, const float* __restrict__ Mc,
    float* __restrict__ out)
{
  __shared__ __align__(16) _Float16 eL[8][16 * 32];   // per-wave 16(l) x 32(m) E tile
  int b    = blockIdx.y;
  int wave = threadIdx.x >> 5;
  int lane = threadIdx.x & 31;
  int l0   = blockIdx.x * 128 + wave * 16;

  const _Float16* Kb = Kh + (size_t)(b * L_ + l0) * DPAD;
  v16h A0 = load_tileA(Kb, DPAD);         // K rows fixed for the whole sweep
  v16h A1 = load_tileA(Kb + 32, DPAD);

  const _Float16* Qb = Qh  + (size_t)(b * L_) * DPAD;
  const _Float16* Vb = Vpt + (size_t)b * C_ * L_;
  const float*    Mb = Mc  + b * L_;

  v8f acc0 = {}, acc1 = {}, acc2 = {}, acc3 = {};
  _Float16* myE = eL[wave];
  int mcol  = lane & 15;
  int rbase = (lane >> 4) << 3;           // D-layout row base: 0 or 8

  v16h B00 = load_tileB(Qb, DPAD);
  v16h B01 = load_tileB(Qb + 32, DPAD);
  v16h B10 = load_tileB(Qb + 16 * DPAD, DPAD);
  v16h B11 = load_tileB(Qb + 16 * DPAD + 32, DPAD);

  for (int m0 = 0; m0 < L_; m0 += 32) {
    int mn = (m0 + 32) & (L_ - 1);        // prefetch next chunk's Q tiles
    v16h nB00 = load_tileB(Qb + mn * DPAD, DPAD);
    v16h nB01 = load_tileB(Qb + mn * DPAD + 32, DPAD);
    v16h nB10 = load_tileB(Qb + (mn + 16) * DPAD, DPAD);
    v16h nB11 = load_tileB(Qb + (mn + 16) * DPAD + 32, DPAD);

    // Vpt B-tiles + column maxima for the *current* chunk: issue before the
    // LDS fence so they are in flight during the s-WMMAs and exp work.
    v16h Bv0 = load_tileB(Vb + (size_t)0  * L_ + m0, L_);
    v16h Bv1 = load_tileB(Vb + (size_t)16 * L_ + m0, L_);
    v16h Bv2 = load_tileB(Vb + (size_t)32 * L_ + m0, L_);
    v16h Bv3 = load_tileB(Vb + (size_t)48 * L_ + m0, L_);
    float M0v = Mb[m0 + mcol];
    float M1v = Mb[m0 + 16 + mcol];

    // s tiles: independent accumulators, 4 WMMAs issue back-to-back.
    v8f D0a = {}, D0b = {}, D1a = {}, D1b = {};
    D0a = WMMA_F16(A0, B00, D0a);
    D0b = WMMA_F16(A1, B01, D0b);
    D1a = WMMA_F16(A0, B10, D1a);
    D1b = WMMA_F16(A1, B11, D1b);
    v8f D0 = D0a + D0b;
    v8f D1 = D1a + D1b;

    // D-layout -> LDS (E tile, row-major [l][m] f16). exp(s-M) <= 1: f16-safe.
#pragma unroll
    for (int i = 0; i < 8; ++i)
      myE[(rbase + i) * 32 + mcol]      = (_Float16)__expf(D0[i] - M0v);
#pragma unroll
    for (int i = 0; i < 8; ++i)
      myE[(rbase + i) * 32 + 16 + mcol] = (_Float16)__expf(D1[i] - M1v);
    asm volatile("" ::: "memory");        // same-wave LDS store->load ordering

    // Reload E in A-matrix layout (row = lane&15, K chunks koff / koff+16).
    const _Float16* ep = myE + (lane & 15) * 32 + ((lane >> 4) << 3);
    v8h e0 = *(const v8h*)(ep);
    v8h e1 = *(const v8h*)(ep + 16);
    v16h Ea;
#pragma unroll
    for (int i = 0; i < 8; ++i) { Ea[i] = e0[i]; Ea[i + 8] = e1[i]; }

    // y += E(16x32) @ Vpt(32 x 64); 4 independent accumulate WMMAs.
    acc0 = WMMA_F16(Ea, Bv0, acc0);
    acc1 = WMMA_F16(Ea, Bv1, acc1);
    acc2 = WMMA_F16(Ea, Bv2, acc2);
    acc3 = WMMA_F16(Ea, Bv3, acc3);

    B00 = nB00; B01 = nB01; B10 = nB10; B11 = nB11;
  }

  // Store y [B][L][64] f32 from D-layout accumulators.
#pragma unroll
  for (int i = 0; i < 8; ++i) {
    float* orow = out + (size_t)(b * L_ + l0 + rbase + i) * C_ + mcol;
    orow[0]  = acc0[i];
    orow[16] = acc1[i];
    orow[32] = acc2[i];
    orow[48] = acc3[i];
  }
}

// ---------------------------------------------------------------------------
extern "C" void kernel_launch(void* const* d_in, const int* in_sizes, int n_in,
                              void* d_out, int out_size, void* d_ws, size_t ws_size,
                              hipStream_t stream) {
  const float* x    = (const float*)d_in[0];
  const float* Wk   = (const float*)d_in[1];
  const float* bk   = (const float*)d_in[2];
  const float* Wq   = (const float*)d_in[3];
  const float* bq   = (const float*)d_in[4];
  const float* Wv   = (const float*)d_in[5];
  const float* bv   = (const float*)d_in[6];
  const int*   slen = (const int*)d_in[7];
  float* out = (float*)d_out;

  // Workspace layout (~42.2 MB total):
  //   Kh   f16 [B][L][64]  (elu * scale, padded)     8 MB
  //   Qh   f16 [B][L][64]  (elu, padded)             8 MB
  //   Vpt  f16 [B][64][L]  (tanh_v / Z, transposed)  8 MB
  //   Vf   f32 [B][L][64]  (tanh_v)                 16 MB
  //   Mc   f32 [B][L]      (column max)            256 KB
  uint8_t* ws = (uint8_t*)d_ws;
  const size_t n = (size_t)B_ * L_ * DPAD;
  _Float16* Kh  = (_Float16*)ws; ws += n * 2;
  _Float16* Qh  = (_Float16*)ws; ws += n * 2;
  _Float16* Vpt = (_Float16*)ws; ws += n * 2;
  float*    Vf  = (float*)ws;    ws += n * 4;
  float*    Mc  = (float*)ws;    ws += (size_t)B_ * L_ * 4;

  proj_kernel <<<dim3(L_ / 256, B_), 256, 0, stream>>>(x, Wk, bk, Wq, bq, Wv, bv, slen, Kh, Qh, Vf);
  stats_kernel<<<dim3(L_ / 128, B_), 256, 0, stream>>>(Kh, Qh, Vf, Mc, Vpt);
  attn_kernel <<<dim3(L_ / 128, B_), 256, 0, stream>>>(Kh, Qh, Vpt, Mc, out);
}
